// RationaleSelectorModel_24532853195259
// MI455X (gfx1250) — compile-verified, matrix operand order checked by reading, not run on hardware
//
#include <hip/hip_runtime.h>
#include <hip/hip_bf16.h>
#include <math.h>

// ---- shapes (fixed by reference setup_inputs) ----
#define B_ 32
#define S_ 4096
#define D_ 768
#define H_ 1024

typedef __attribute__((ext_vector_type(16))) __bf16 v16bf;
typedef __attribute__((ext_vector_type(8)))  __bf16 bf16x8;
typedef __attribute__((ext_vector_type(8)))  float  v8f;

static constexpr int AW = D_ + 8;   // padded LDS row stride (bf16 elems) to dodge bank conflicts

// -------------------------------------------------------------------------
// Kernel 0: convert w1 (f32, [H,D]) -> bf16 in workspace (L2-resident, 1.5MB)
// -------------------------------------------------------------------------
__global__ void convert_w1_bf16(const float* __restrict__ w1,
                                __bf16* __restrict__ w1b, int n) {
  int i = blockIdx.x * blockDim.x + threadIdx.x;
  if (i < n) w1b[i] = (__bf16)w1[i];
}

// -------------------------------------------------------------------------
// Kernel 1: fused  mask -> LayerNorm -> (x @ w1^T) -> GELU -> (· w2) + b2
// 32 tokens per workgroup, 256 threads (8 waves, wave32).
// wave w owns H slice [w*128, w*128+128) and BOTH 16-token M-tiles, so each
// B fragment feeds two independent WMMA accumulator chains (accA / accB):
// w1 is read exactly once per workgroup and the matrix pipe gets two
// independent chains to interleave.
// -------------------------------------------------------------------------
__global__ __launch_bounds__(256)
void scorer_kernel(const float* __restrict__ emb,
                   const float* __restrict__ attn,
                   const float* __restrict__ ln_g,
                   const float* __restrict__ ln_b,
                   const __bf16* __restrict__ w1b,
                   const float* __restrict__ b1,
                   const float* __restrict__ w2,
                   const float* __restrict__ b2,
                   float* __restrict__ scores) {
  __shared__ __align__(16) __bf16 aTile[32][AW];
  __shared__ float scoreAcc[32];

  const int tid    = threadIdx.x;
  const int wave   = tid >> 5;
  const int lane   = tid & 31;
  const int laneLo = lane & 15;
  const int hiHalf = lane >> 4;
  const long tokBase = (long)blockIdx.x * 32;

  if (tid < 32) scoreAcc[tid] = 0.f;

  // ---- Phase 1: masked LayerNorm, write bf16 A-tile. wave handles 4 tokens.
  for (int tt = 0; tt < 4; ++tt) {
    const int  t   = wave * 4 + tt;
    const long tok = tokBase + t;
    const float am = attn[tok];
    const float* row = emb + tok * (long)D_;
    float4 v[6];
    float sum = 0.f, ssq = 0.f;
#pragma unroll
    for (int j = 0; j < 6; ++j) {
      const int d0 = j * 128 + lane * 4;       // 768 = 6 * 128
      float4 e = *(const float4*)(row + d0);
      e.x *= am; e.y *= am; e.z *= am; e.w *= am;
      v[j] = e;
      sum += e.x + e.y + e.z + e.w;
      ssq += e.x * e.x + e.y * e.y + e.z * e.z + e.w * e.w;
    }
#pragma unroll
    for (int off = 16; off >= 1; off >>= 1) {
      sum += __shfl_xor(sum, off, 32);
      ssq += __shfl_xor(ssq, off, 32);
    }
    const float mu   = sum * (1.f / D_);
    const float var  = ssq * (1.f / D_) - mu * mu;
    const float rstd = rsqrtf(var + 1e-5f);
#pragma unroll
    for (int j = 0; j < 6; ++j) {
      const int d0 = j * 128 + lane * 4;
      const float4 g  = *(const float4*)(ln_g + d0);
      const float4 bb = *(const float4*)(ln_b + d0);
      __bf16* dst = &aTile[t][d0];
      dst[0] = (__bf16)((v[j].x - mu) * rstd * g.x + bb.x);
      dst[1] = (__bf16)((v[j].y - mu) * rstd * g.y + bb.y);
      dst[2] = (__bf16)((v[j].z - mu) * rstd * g.z + bb.z);
      dst[3] = (__bf16)((v[j].w - mu) * rstd * g.w + bb.w);
    }
  }
  __syncthreads();

  // ---- Phase 2: WMMA GEMM + GELU + w2 contraction
  const int h0 = wave * 128;                    // 8 waves x 128 = H
  const __bf16* aRow0 = aTile[laneLo];          // M-tile 0 (tokens 0..15)
  const __bf16* aRow1 = aTile[16 + laneLo];     // M-tile 1 (tokens 16..31)

  float sp0[8], sp1[8];
#pragma unroll
  for (int r = 0; r < 8; ++r) { sp0[r] = 0.f; sp1[r] = 0.f; }

  for (int nt = 0; nt < 8; ++nt) {
    const int n = h0 + nt * 16 + laneLo;
    const __bf16* bRow = w1b + (long)n * D_;
    if (nt + 1 < 8)
      __builtin_prefetch(w1b + (long)(n + 16) * D_, 0, 1);  // global_prefetch_b8

    v8f accA = {};
    v8f accB = {};
#pragma unroll 4
    for (int k0 = 0; k0 < D_; k0 += 32) {
      const int kb = k0 + hiHalf * 8;          // ISA 16-bit A/B lane-half K split
      v16bf a0, a1, bfrag;
      const bf16x8 a0lo = *(const bf16x8*)(aRow0 + kb);
      const bf16x8 a0hi = *(const bf16x8*)(aRow0 + kb + 16);
      const bf16x8 a1lo = *(const bf16x8*)(aRow1 + kb);
      const bf16x8 a1hi = *(const bf16x8*)(aRow1 + kb + 16);
      const bf16x8 blo  = *(const bf16x8*)(bRow  + kb);
      const bf16x8 bhi  = *(const bf16x8*)(bRow  + kb + 16);
#pragma unroll
      for (int e = 0; e < 8; ++e) {
        a0[e] = a0lo[e]; a0[e + 8] = a0hi[e];
        a1[e] = a1lo[e]; a1[e + 8] = a1hi[e];
        bfrag[e] = blo[e]; bfrag[e + 8] = bhi[e];
      }
      accA = __builtin_amdgcn_wmma_f32_16x16x32_bf16(false, a0, false, bfrag,
                                                     (short)0, accA, false, false);
      accB = __builtin_amdgcn_wmma_f32_16x16x32_bf16(false, a1, false, bfrag,
                                                     (short)0, accB, false, false);
    }
    // acc[r]: lane<16 -> token r (+16 for accB), lane>=16 -> token 8+r; column n
    const float bias = b1[n];
    const float wv   = w2[n];
#pragma unroll
    for (int r = 0; r < 8; ++r) {
      const float cA = accA[r] + bias;
      const float cB = accB[r] + bias;
      sp0[r] += 0.5f * cA * (1.f + erff(cA * 0.70710678118f)) * wv;  // exact GELU
      sp1[r] += 0.5f * cB * (1.f + erff(cB * 0.70710678118f)) * wv;
    }
  }
  // reduce the 16 columns held across lanes of each half-wave
#pragma unroll
  for (int r = 0; r < 8; ++r) {
    float s0 = sp0[r], s1 = sp1[r];
    s0 += __shfl_xor(s0, 1, 32); s1 += __shfl_xor(s1, 1, 32);
    s0 += __shfl_xor(s0, 2, 32); s1 += __shfl_xor(s1, 2, 32);
    s0 += __shfl_xor(s0, 4, 32); s1 += __shfl_xor(s1, 4, 32);
    s0 += __shfl_xor(s0, 8, 32); s1 += __shfl_xor(s1, 8, 32);
    if (laneLo == 0) {
      atomicAdd(&scoreAcc[hiHalf * 8 + r], s0);        // ds_add_f32
      atomicAdd(&scoreAcc[16 + hiHalf * 8 + r], s1);
    }
  }
  __syncthreads();

  if (tid < 32) {
    const long tok = tokBase + tid;
    float sc = scoreAcc[tid] + b2[0];
    if (attn[tok] == 0.f) sc = -1e9f;
    scores[tok] = sc;
  }
}

// -------------------------------------------------------------------------
// Kernel 2: per-batch-row entmax-1.5 over S + Kumaraswamy gate.
// One 1024-thread block per batch row. Score row staged once into LDS via
// async global->LDS copy (ASYNCcnt), then bitonic sort + block scan in LDS.
// out layout: z [B*S] | g [B*S] | reg [1]
// -------------------------------------------------------------------------
__device__ __forceinline__ float softplusf(float v) {
  if (v > 20.f)  return v;
  if (v < -20.f) return expf(v);
  return log1pf(expf(v));
}

__global__ __launch_bounds__(1024)
void entmax_kernel(const float* __restrict__ scores,
                   const float* __restrict__ attn,
                   const float* __restrict__ u,
                   float* __restrict__ out) {
  __shared__ __align__(16) float raw[S_];   // raw score row (async-staged)
  __shared__ float sm[S_];                  // sorted x
  __shared__ float tauArr[S_];              // tau per k
  __shared__ float cs1[1024];
  __shared__ float cs2[1024];
  __shared__ float red[32];
  __shared__ float sMax;
  __shared__ float sTau;
  __shared__ int   sSupport;

  const int tid  = threadIdx.x;
  const int b    = blockIdx.x;
  const int lane = tid & 31;
  const int wv   = tid >> 5;
  const float* srow = scores + (long)b * S_;

  // ---- stage score row into LDS with CDNA5 async tensor/LDS path
  {
    const float* gsrc = srow + tid * 4;                     // 16B per thread
    const unsigned ldsOff = (unsigned)(size_t)(&raw[tid * 4]);
    asm volatile("global_load_async_to_lds_b128 %0, %1, off"
                 :: "v"(ldsOff), "v"(gsrc) : "memory");
    asm volatile("s_wait_asynccnt 0x0" ::: "memory");
  }
  __syncthreads();

  // x = (scores / TAU) / 2, minus row max
  float x[4];
  float mx = -__builtin_inff();
#pragma unroll
  for (int j = 0; j < 4; ++j) {
    x[j] = raw[tid + j * 1024] * 0.5f;   // TAU = 1
    mx = fmaxf(mx, x[j]);
  }
#pragma unroll
  for (int off = 16; off >= 1; off >>= 1) mx = fmaxf(mx, __shfl_xor(mx, off, 32));
  if (lane == 0) red[wv] = mx;
  __syncthreads();
  if (tid == 0) {
    float m = red[0];
    for (int i = 1; i < 32; ++i) m = fmaxf(m, red[i]);
    sMax = m;
  }
  __syncthreads();
  const float xmax = sMax;
#pragma unroll
  for (int j = 0; j < 4; ++j) {
    x[j] -= xmax;
    sm[tid + j * 1024] = x[j];
  }
  __syncthreads();

  // bitonic sort, descending
  for (int k = 2; k <= S_; k <<= 1) {
    for (int j = k >> 1; j > 0; j >>= 1) {
#pragma unroll
      for (int rep = 0; rep < 4; ++rep) {
        const int i = tid + rep * 1024;
        const int l = i ^ j;
        if (l > i) {
          const float a  = sm[i];
          const float bb = sm[l];
          const bool desc = ((i & k) == 0);
          if (desc ? (a < bb) : (a > bb)) { sm[i] = bb; sm[l] = a; }
        }
      }
      __syncthreads();
    }
  }

  // cumsum and cumsum of squares: per-thread chunk of 4 + block scan of chunk sums
  float p1[4], p2[4];
  {
    float r1 = 0.f, r2 = 0.f;
#pragma unroll
    for (int jj = 0; jj < 4; ++jj) {
      const float vvv = sm[tid * 4 + jj];
      r1 += vvv; r2 += vvv * vvv;
      p1[jj] = r1; p2[jj] = r2;
    }
    cs1[tid] = r1; cs2[tid] = r2;
  }
  __syncthreads();
  for (int off = 1; off < 1024; off <<= 1) {
    float a1 = cs1[tid], a2 = cs2[tid];
    if (tid >= off) { a1 += cs1[tid - off]; a2 += cs2[tid - off]; }
    __syncthreads();
    cs1[tid] = a1; cs2[tid] = a2;
    __syncthreads();
  }
  const float o1 = (tid > 0) ? cs1[tid - 1] : 0.f;
  const float o2 = (tid > 0) ? cs2[tid - 1] : 0.f;

  // tau(k) and support size
  int flags = 0;
#pragma unroll
  for (int jj = 0; jj < 4; ++jj) {
    const int   k    = tid * 4 + jj + 1;     // 1-based rank
    const float rho  = (float)k;
    const float mean = (o1 + p1[jj]) / rho;
    const float msq  = (o2 + p2[jj]) / rho;
    const float ssv  = rho * (msq - mean * mean);
    float delta = (1.f - ssv) / rho;
    delta = fmaxf(delta, 0.f);
    const float tau = mean - sqrtf(delta);
    tauArr[k - 1] = tau;
    flags += (tau <= sm[k - 1]) ? 1 : 0;
  }
#pragma unroll
  for (int off = 16; off >= 1; off >>= 1) flags += __shfl_xor(flags, off, 32);
  if (lane == 0) red[wv] = (float)flags;
  __syncthreads();
  if (tid == 0) {
    int tot = 0;
    for (int i = 0; i < 32; ++i) tot += (int)red[i];
    sSupport = tot;
    sTau = tauArr[tot - 1];
  }
  __syncthreads();
  const float tauStar = sTau;

  // z = clip(x - tau*, 0)^2 * attn ; Kumaraswamy straight-through gate g = h
  const long base = (long)b * S_;
#pragma unroll
  for (int j = 0; j < 4; ++j) {
    const long i  = base + tid + j * 1024;
    const float am = attn[i];
    float t = fmaxf(x[j] - tauStar, 0.f);
    const float z = t * t * am;
    out[i] = z;

    const float sraw = raw[tid + j * 1024];               // from LDS stage
    const float eff  = sraw + 2.0f * (2.f * z - 1.f);     // COUPLING_K = 2
    const float a    = softplusf(eff) + 1e-6f;
    const float bb   = softplusf(-eff) + 1e-6f;
    float uc = fminf(fmaxf(u[i], 1e-6f), 1.f - 1e-6f);
    const float pb = powf(1.f - uc, 1.f / bb);
    const float xk = powf(fmaxf(1.f - pb, 0.f), 1.f / a);
    const float y  = -0.1f + 1.2f * xk;                   // L + (R-L)*xk
    const float yc = fminf(fmaxf(y, 0.f), 1.f);
    out[(long)B_ * S_ + i] = (yc > 0.5f) ? am : 0.f;      // g == h forward value
  }
  if (b == 0 && tid == 0) out[2L * B_ * S_] = 0.f;        // reg
}

// -------------------------------------------------------------------------
extern "C" void kernel_launch(void* const* d_in, const int* in_sizes, int n_in,
                              void* d_out, int out_size, void* d_ws, size_t ws_size,
                              hipStream_t stream) {
  const float* emb  = (const float*)d_in[0];
  const float* attn = (const float*)d_in[1];
  const float* u    = (const float*)d_in[2];
  const float* ln_g = (const float*)d_in[3];
  const float* ln_b = (const float*)d_in[4];
  const float* w1   = (const float*)d_in[5];
  const float* b1   = (const float*)d_in[6];
  const float* w2   = (const float*)d_in[7];
  const float* b2   = (const float*)d_in[8];
  float* out = (float*)d_out;

  char* ws = (char*)d_ws;
  __bf16* w1b = (__bf16*)ws;                                   // H*D bf16 = 1.5 MB
  const size_t w1bBytes = ((size_t)H_ * D_ * sizeof(__bf16) + 255) & ~(size_t)255;
  float* scores = (float*)(ws + w1bBytes);                     // B*S f32 = 0.5 MB

  convert_w1_bf16<<<(H_ * D_ + 255) / 256, 256, 0, stream>>>(w1, w1b, H_ * D_);
  scorer_kernel<<<(B_ * S_) / 32, 256, 0, stream>>>(emb, attn, ln_g, ln_b,
                                                    w1b, b1, w2, b2, scores);
  entmax_kernel<<<B_, 1024, 0, stream>>>(scores, attn, u, out);
}
